// ArrowAttention_79087527788637
// MI455X (gfx1250) — matrix operand order repaired and checked by reading, not compiled, not run on hardware
//
#include <hip/hip_runtime.h>
#include <hip/hip_bf16.h>

// B=2, S=2048, E=1024, H=16, D=64, WIN=16
// M = B*S = 4096, 3E = 3072, SQ = 2047, B*SQ = 4094

typedef unsigned short u16;
typedef __attribute__((ext_vector_type(16))) __bf16          v16bf;
typedef __attribute__((ext_vector_type(8)))  float           v8f;
typedef __attribute__((ext_vector_type(8)))  unsigned short  ush8;
typedef __attribute__((ext_vector_type(4)))  int             v4i;

#if defined(__has_builtin)
#if __has_builtin(__builtin_amdgcn_global_load_async_to_lds_b128)
#define HAVE_ASYNC_LDS 1
#endif
#endif

#if defined(HAVE_ASYNC_LDS)
typedef __attribute__((address_space(1))) v4i gv4i;   // global 16B chunk
typedef __attribute__((address_space(3))) v4i lv4i;   // LDS    16B chunk
#if __has_builtin(__builtin_amdgcn_s_wait_asynccnt)
#define WAIT_ASYNC(n) __builtin_amdgcn_s_wait_asynccnt(n)
#else
#define WAIT_ASYNC(n) asm volatile("s_wait_asynccnt %0" ::"i"(n))
#endif
#endif

static __device__ __forceinline__ float bf2f(u16 b) {
    return __uint_as_float(((unsigned)b) << 16);
}
static __device__ __forceinline__ u16 f2bf(float f) {
    unsigned u = __float_as_uint(f);
    return (u16)((u + 0x7fffu + ((u >> 16) & 1u)) >> 16);   // round-to-nearest-even
}

// ---------------------------------------------------------------- fp32 -> bf16
__global__ __launch_bounds__(256) void cvt_f32_bf16(const float* __restrict__ s,
                                                    u16* __restrict__ d, int n) {
    for (int i = blockIdx.x * blockDim.x + threadIdx.x; i < n;
         i += gridDim.x * blockDim.x)
        d[i] = f2bf(s[i]);
}

// ---------------------------------------------------------------- WMMA GEMM
// C(M,N) = A(M,K) @ W(N,K)^T + bias(N).  A, W bf16 (K-contiguous), acc fp32.
// Block: 256 threads = 8 waves, tile 64(M) x 128(N); wave tile 16 x 64.
// Main path: double-buffered LDS staging via GLOBAL_LOAD_ASYNC_TO_LDS_B128
// (ASYNCcnt), fragments read back with ds_load_b128.
// Fallback: register double-buffered direct global loads.
// Ch != nullptr : store bf16, plain row-major (row stride N).
// Ch == nullptr : store fp32 into Cf with remap:
//   remap 0: row m            -> row m
//   remap 1: m = b*2047+(q-1) -> row b*2048 + q   (local tokens into d_out)
//   remap 2: m = b            -> row b*2048       (cls token into d_out)
__global__ __launch_bounds__(256) void gemm_bf16_wmma(
    const u16* __restrict__ A, const u16* __restrict__ W,
    const float* __restrict__ bias, float* __restrict__ Cf,
    u16* __restrict__ Ch, int M, int N, int K, int remap)
{
    const int tid  = threadIdx.x;
    const int lane = tid & 31;
    const int wave = tid >> 5;
    const int lo   = lane & 15;
    const int hi   = lane >> 4;
    const int tileN = blockIdx.x * 128;
    const int tileM = blockIdx.y * 64;
    const int wMl   = (wave >> 1) * 16;     // wave tile origin within block tile
    const int wNl   = (wave & 1) * 64;

    v8f acc[4] = {};
    union Frag { v16bf v; ush8 h[2]; };

#if defined(HAVE_ASYNC_LDS)
    // ---- LDS-staged, ASYNCcnt double-buffered pipeline ----
    __shared__ u16 As[2][64 * 32];      //  8 KB: 64 rows x 32 K
    __shared__ u16 Bs[2][128 * 32];     // 16 KB: 128 rows x 32 K

    // Per-thread copy assignments (16B chunks): A = 256 chunks, B = 512 chunks.
    const int crow = tid >> 2;          // 0..63
    const int cchk = tid & 3;           // 0..3
    int arowg = tileM + crow; if (arowg >= M) arowg = M - 1;   // clamp OOB rows
    const u16* agp  = A + (size_t)arowg * K + cchk * 8;
    const u16* bgp0 = W + (size_t)(tileN + crow) * K + cchk * 8;
    const u16* bgp1 = W + (size_t)(tileN + 64 + crow) * K + cchk * 8;
    u16* alp  = &As[0][crow * 32 + cchk * 8];
    u16* blp0 = &Bs[0][crow * 32 + cchk * 8];
    u16* blp1 = &Bs[0][(64 + crow) * 32 + cchk * 8];
    const int bufA = 64 * 32, bufB = 128 * 32;

    auto stage = [&](int buf, int k) {
        __builtin_amdgcn_global_load_async_to_lds_b128(
            (gv4i*)(agp + k),  (lv4i*)(alp  + buf * bufA), 0, 0);
        __builtin_amdgcn_global_load_async_to_lds_b128(
            (gv4i*)(bgp0 + k), (lv4i*)(blp0 + buf * bufB), 0, 0);
        __builtin_amdgcn_global_load_async_to_lds_b128(
            (gv4i*)(bgp1 + k), (lv4i*)(blp1 + buf * bufB), 0, 0);
    };

    const int nsteps = K >> 5;
    stage(0, 0);                                   // ASYNCcnt = 3
    for (int i = 0; i < nsteps; ++i) {
        const int cur = i & 1;
        if (i + 1 < nsteps) {
            stage(cur ^ 1, (i + 1) * 32);          // ASYNCcnt = 6
            WAIT_ASYNC(3);                         // oldest stage complete
        } else {
            WAIT_ASYNC(0);
        }
        __syncthreads();                           // all waves' copies visible

        Frag a, bf[4];
        const u16* Al = &As[cur][(wMl + lo) * 32 + hi * 8];
        a.h[0] = *(const ush8*)(Al);
        a.h[1] = *(const ush8*)(Al + 16);
#pragma unroll
        for (int j = 0; j < 4; ++j) {
            const u16* Bl = &Bs[cur][(wNl + 16 * j + lo) * 32 + hi * 8];
            bf[j].h[0] = *(const ush8*)(Bl);
            bf[j].h[1] = *(const ush8*)(Bl + 16);
        }
#pragma unroll
        for (int j = 0; j < 4; ++j)
            acc[j] = __builtin_amdgcn_wmma_f32_16x16x32_bf16(
                false, a.v, false, bf[j].v, (short)0, acc[j], false, false);

        __syncthreads();     // reads done before this buffer is rewritten
    }
#else
    // ---- fallback: register double-buffered direct global loads ----
    int rowA = tileM + wMl + lo;
    if (rowA >= M) rowA = M - 1;
    const u16* Ap = A + (size_t)rowA * K + hi * 8;
    const u16* Bp[4];
#pragma unroll
    for (int j = 0; j < 4; ++j)
        Bp[j] = W + (size_t)(tileN + wNl + 16 * j + lo) * K + hi * 8;

    Frag a[2], b[2][4];
    auto loadA = [&](Frag& f, int k) {
        f.h[0] = *(const ush8*)(Ap + k);
        f.h[1] = *(const ush8*)(Ap + k + 16);
    };
    auto loadB = [&](Frag& f, const u16* p, int k) {
        f.h[0] = *(const ush8*)(p + k);
        f.h[1] = *(const ush8*)(p + k + 16);
    };
    loadA(a[0], 0);
#pragma unroll
    for (int j = 0; j < 4; ++j) loadB(b[0][j], Bp[j], 0);

    int cur = 0;
    for (int k0 = 0; k0 < K; k0 += 32, cur ^= 1) {
        if (k0 + 32 < K) {
            loadA(a[cur ^ 1], k0 + 32);
            __builtin_prefetch(Ap + k0 + 160, 0, 1);
#pragma unroll
            for (int j = 0; j < 4; ++j) loadB(b[cur ^ 1][j], Bp[j], k0 + 32);
        }
#pragma unroll
        for (int j = 0; j < 4; ++j)
            acc[j] = __builtin_amdgcn_wmma_f32_16x16x32_bf16(
                false, a[cur].v, false, b[cur][j].v, (short)0, acc[j], false, false);
    }
#endif

    float bv[4];
#pragma unroll
    for (int j = 0; j < 4; ++j) bv[j] = bias[tileN + wNl + 16 * j + lo];

    // C/D layout: vgpr r, lane -> m = base + r + hi*8, n = col base + lo
#pragma unroll
    for (int r = 0; r < 8; ++r) {
        int m = tileM + wMl + r + hi * 8;
        if (m >= M) continue;
#pragma unroll
        for (int j = 0; j < 4; ++j) {
            int   n   = tileN + wNl + 16 * j + lo;
            float out = acc[j][r] + bv[j];
            if (Ch) {
                Ch[(size_t)m * N + n] = f2bf(out);
            } else {
                size_t orow;
                if (remap == 1)      orow = (size_t)(m / 2047) * 2048 + 1 + (m % 2047);
                else if (remap == 2) orow = (size_t)m * 2048;
                else                 orow = (size_t)m;
                Cf[orow * (size_t)N + n] = out;
            }
        }
    }
}

// ------------------------------------------------- global (cls) attention
// One block per (b,h). qkv: (4096, 3072) bf16; q|k|v at col 0|1024|2048.
// ctx: (2, 1024) bf16.
__global__ __launch_bounds__(256) void attn_global(const u16* __restrict__ qkv,
                                                   u16* __restrict__ ctx)
{
    __shared__ float qs[64];
    __shared__ float red[256];
    __shared__ float p[2048];
    const int b   = blockIdx.x >> 4;
    const int h   = blockIdx.x & 15;
    const int tid = threadIdx.x;
    const u16* base = qkv + (size_t)b * 2048 * 3072;

    if (tid < 64) qs[tid] = bf2f(base[h * 64 + tid]) * 0.125f;   // q(token0) * 1/sqrt(64)
    __syncthreads();

    float lmax = -3e38f;
    for (int j = tid; j < 2048; j += 256) {
        const u16* kr = base + (size_t)j * 3072 + 1024 + h * 64;
        float s = 0.f;
#pragma unroll
        for (int d = 0; d < 64; ++d) s += qs[d] * bf2f(kr[d]);
        p[j] = s;
        lmax = fmaxf(lmax, s);
    }
    red[tid] = lmax; __syncthreads();
    for (int off = 128; off > 0; off >>= 1) {
        if (tid < off) red[tid] = fmaxf(red[tid], red[tid + off]);
        __syncthreads();
    }
    const float mx = red[0];
    __syncthreads();

    float lsum = 0.f;
    for (int j = tid; j < 2048; j += 256) {
        float e = __expf(p[j] - mx);
        p[j] = e; lsum += e;
    }
    red[tid] = lsum; __syncthreads();
    for (int off = 128; off > 0; off >>= 1) {
        if (tid < off) red[tid] += red[tid + off];
        __syncthreads();
    }
    const float inv = 1.f / red[0];
    __syncthreads();

    const int d = tid & 63, g = tid >> 6;
    float acc = 0.f;
    for (int j = g; j < 2048; j += 4)
        acc += p[j] * bf2f(base[(size_t)j * 3072 + 2048 + h * 64 + d]);
    red[tid] = acc; __syncthreads();
    if (tid < 64) {
        float c = (red[tid] + red[tid + 64] + red[tid + 128] + red[tid + 192]) * inv;
        ctx[(size_t)b * 1024 + h * 64 + tid] = f2bf(c);
    }
}

// ------------------------------------------------- local (windowed) attention
// One wave32 per (b, q, h); window w=0..32 -> kidx = q-16+w, valid iff 1<=kidx<2048.
// ctx: (4094, 1024) bf16, row = b*2047 + (q-1).
__global__ __launch_bounds__(256) void attn_local(const u16* __restrict__ qkv,
                                                  u16* __restrict__ ctx)
{
    const int lane = threadIdx.x & 31;
    const int wid  = blockIdx.x * 8 + (threadIdx.x >> 5);
    if (wid >= 2 * 2047 * 16) return;
    const int h = wid & 15;
    const int t = wid >> 4;
    const int q = (t % 2047) + 1;
    const int b = t / 2047;

    const u16* base = qkv + (size_t)b * 2048 * 3072;
    const u16* qrow = base + (size_t)q * 3072 + h * 64;
    float qv[64];
#pragma unroll
    for (int d = 0; d < 64; ++d) qv[d] = bf2f(qrow[d]) * 0.125f;

    auto score = [&](int kidx) -> float {
        int kc = kidx < 0 ? 0 : (kidx > 2047 ? 2047 : kidx);
        const u16* kr = base + (size_t)kc * 3072 + 1024 + h * 64;
        float s = 0.f;
#pragma unroll
        for (int d = 0; d < 64; ++d) s += qv[d] * bf2f(kr[d]);
        return (kidx >= 1 && kidx < 2048) ? s : -1e30f;
    };

    float s1 = score(q - 16 + lane);                       // w = lane
    float s2 = (lane == 0) ? score(q + 16) : -1e30f;       // w = 32 (lane 0 only)
    float m = fmaxf(s1, s2);
#pragma unroll
    for (int off = 16; off > 0; off >>= 1) m = fmaxf(m, __shfl_xor(m, off, 32));
    float p1 = __expf(s1 - m);
    float p2 = (lane == 0) ? __expf(s2 - m) : 0.f;
    float sum = p1 + p2;
#pragma unroll
    for (int off = 16; off > 0; off >>= 1) sum += __shfl_xor(sum, off, 32);
    const float inv = 1.f / sum;

    float a0 = 0.f, a1 = 0.f;                              // dims lane, lane+32
    for (int w = 0; w < 33; ++w) {
        float pw = (w < 32) ? __shfl(p1, w, 32) : __shfl(p2, 0, 32);
        int kidx = q - 16 + w;
        int kc   = kidx < 0 ? 0 : (kidx > 2047 ? 2047 : kidx);
        const u16* vr = base + (size_t)kc * 3072 + 2048 + h * 64;
        a0 += pw * bf2f(vr[lane]);
        a1 += pw * bf2f(vr[lane + 32]);
    }
    u16* cr = ctx + (size_t)(b * 2047 + (q - 1)) * 1024 + h * 64;
    cr[lane]      = f2bf(a0 * inv);
    cr[lane + 32] = f2bf(a1 * inv);
}

// ---------------------------------------------------------------- launch
extern "C" void kernel_launch(void* const* d_in, const int* in_sizes, int n_in,
                              void* d_out, int out_size, void* d_ws, size_t ws_size,
                              hipStream_t stream)
{
    (void)in_sizes; (void)n_in; (void)out_size; (void)ws_size;
    const float* hs        = (const float*)d_in[0];
    const float* in_w_cls  = (const float*)d_in[1];
    const float* in_b_cls  = (const float*)d_in[2];
    const float* out_w_cls = (const float*)d_in[3];
    const float* out_b_cls = (const float*)d_in[4];
    const float* in_w_loc  = (const float*)d_in[5];
    const float* in_b_loc  = (const float*)d_in[6];
    const float* out_w_loc = (const float*)d_in[7];
    const float* out_b_loc = (const float*)d_in[8];
    float* out = (float*)d_out;

    // workspace layout (total = 80 MiB)
    char* w = (char*)d_ws;
    u16* hbf    = (u16*)w; w += (size_t)4096 * 1024 * 2;   // hidden bf16
    u16* wclsb  = (u16*)w; w += (size_t)3072 * 1024 * 2;   // in_w_cls bf16
    u16* wlocb  = (u16*)w; w += (size_t)3072 * 1024 * 2;   // in_w_loc bf16
    u16* owclsb = (u16*)w; w += (size_t)1024 * 1024 * 2;   // out_w_cls bf16
    u16* owlocb = (u16*)w; w += (size_t)1024 * 1024 * 2;   // out_w_loc bf16
    u16* qkvc   = (u16*)w; w += (size_t)4096 * 3072 * 2;   // cls QKV bf16
    u16* qkvl   = (u16*)w; w += (size_t)4096 * 3072 * 2;   // loc QKV bf16
    u16* ctxc   = (u16*)w; w += (size_t)2    * 1024 * 2;   // cls ctx bf16
    u16* ctxl   = (u16*)w;                                  // loc ctx bf16 (4094x1024)

    cvt_f32_bf16<<<2048, 256, 0, stream>>>(hs,        hbf,    4096 * 1024);
    cvt_f32_bf16<<<2048, 256, 0, stream>>>(in_w_cls,  wclsb,  3072 * 1024);
    cvt_f32_bf16<<<2048, 256, 0, stream>>>(in_w_loc,  wlocb,  3072 * 1024);
    cvt_f32_bf16<<<1024, 256, 0, stream>>>(out_w_cls, owclsb, 1024 * 1024);
    cvt_f32_bf16<<<1024, 256, 0, stream>>>(out_w_loc, owlocb, 1024 * 1024);

    // QKV projections: (4096,1024) @ (3072,1024)^T -> bf16 (4096,3072)
    gemm_bf16_wmma<<<dim3(24, 64), 256, 0, stream>>>(hbf, wclsb, in_b_cls,
                                                     nullptr, qkvc, 4096, 3072, 1024, 0);
    gemm_bf16_wmma<<<dim3(24, 64), 256, 0, stream>>>(hbf, wlocb, in_b_loc,
                                                     nullptr, qkvl, 4096, 3072, 1024, 0);

    attn_global<<<32,   256, 0, stream>>>(qkvc, ctxc);
    attn_local <<<8188, 256, 0, stream>>>(qkvl, ctxl);

    // Output projections straight into d_out (fp32) with row remapping.
    gemm_bf16_wmma<<<dim3(8, 1),  256, 0, stream>>>(ctxc, owclsb, out_b_cls,
                                                    out, nullptr, 2,    1024, 1024, 2);
    gemm_bf16_wmma<<<dim3(8, 64), 256, 0, stream>>>(ctxl, owlocb, out_b_loc,
                                                    out, nullptr, 4094, 1024, 1024, 1);
}
